// TreeDecoder_12635793785520
// MI455X (gfx1250) — matrix operand order repaired and checked by reading, not compiled, not run on hardware
//
#include <hip/hip_runtime.h>

typedef __attribute__((ext_vector_type(16))) __bf16 v16bf;
typedef __attribute__((ext_vector_type(8)))  float  v8f;

#define DD   512
#define TT   64
#define BB   32
#define LXX  64
#define VOC  32500
#define VPAD 32512

__device__ __forceinline__ unsigned short f2bf(float f) {
  unsigned int u = __float_as_uint(f);
  u += 0x7FFFu + ((u >> 16) & 1u);          // round-to-nearest-even
  return (unsigned short)(u >> 16);
}

__device__ __forceinline__ float sigm(float x) { return 1.0f / (1.0f + __expf(-x)); }

// ---------------------------------------------------------------------------
// Wave-level 32x16 GEMM tile: out[m0+0..31][n0..n0+15] += sum_k A[m][k]*W[n][k]
// A, W are bf16 row-major. Uses v_wmma_f32_16x16x32_bf16, two M-tiles per wave.
// ---------------------------------------------------------------------------
__device__ __forceinline__ void wmma_tile32(const unsigned short* A, int lda, int m0,
                                            const unsigned short* W, int ldw, int n0,
                                            int K, v8f& acc0, v8f& acc1)
{
  const int lane = threadIdx.x & 31;
  const int half = lane >> 4;
  const int lrow = lane & 15;
  const __bf16* Ab = reinterpret_cast<const __bf16*>(A);
  const __bf16* Wb = reinterpret_cast<const __bf16*>(W);
  for (int k = 0; k < K; k += 32) {
    // B operand: column (n0+lrow) of W^T == row (n0+lrow) of W, table K order (same as A)
    v16bf bm;
    const __bf16* wr = Wb + (size_t)(n0 + lrow) * ldw + k + half * 8;
#pragma unroll
    for (int j = 0; j < 8; ++j) { bm[j] = wr[j]; bm[8 + j] = wr[16 + j]; }
    // A operand, M-tile 0 (rows m0..m0+15)
    v16bf am;
    const __bf16* a0 = Ab + (size_t)(m0 + lrow) * lda + k + half * 8;
#pragma unroll
    for (int j = 0; j < 8; ++j) { am[j] = a0[j]; am[8 + j] = a0[16 + j]; }
    acc0 = __builtin_amdgcn_wmma_f32_16x16x32_bf16(false, am, false, bm, (short)0, acc0, false, false);
    // A operand, M-tile 1 (rows m0+16..m0+31)
    const __bf16* a1 = Ab + (size_t)(m0 + 16 + lrow) * lda + k + half * 8;
#pragma unroll
    for (int j = 0; j < 8; ++j) { am[j] = a1[j]; am[8 + j] = a1[16 + j]; }
    acc1 = __builtin_amdgcn_wmma_f32_16x16x32_bf16(false, am, false, bm, (short)0, acc1, false, false);
  }
}

// ---------------------------------------------------------------------------
// Utility kernels
// ---------------------------------------------------------------------------
__global__ void cvt_bf16_pad(const float* __restrict__ src, unsigned short* __restrict__ dst,
                             long nsrc, long ndst) {
  long i = (long)blockIdx.x * blockDim.x + threadIdx.x;
  long stride = (long)gridDim.x * blockDim.x;
  for (; i < ndst; i += stride) dst[i] = f2bf(i < nsrc ? src[i] : 0.0f);
}

__global__ void zero_f32(float* p, long n) {
  long i = (long)blockIdx.x * blockDim.x + threadIdx.x;
  long stride = (long)gridDim.x * blockDim.x;
  for (; i < n; i += stride) p[i] = 0.0f;
}

__global__ void copy_hc(const float* __restrict__ src, float* __restrict__ dstf,
                        unsigned short* dstb, int n) {
  int i = blockIdx.x * blockDim.x + threadIdx.x;
  if (i < n) {
    float v = src[i];
    dstf[i] = v;
    if (dstb) dstb[i] = f2bf(v);
  }
}

// ---------------------------------------------------------------------------
// Per-step input assembly: word_in = [emb[tok] | states[par] | wfeed] (bf16)
//                          rule_in[:2048] = [emb[tok] | states[par] | rfeed]
// ---------------------------------------------------------------------------
__global__ void prep_step(const int* __restrict__ ytr, const float* __restrict__ emb,
                          const float* __restrict__ states,
                          const float* __restrict__ rfeed, const float* __restrict__ wfeed,
                          unsigned short* __restrict__ word_in,
                          unsigned short* __restrict__ rule_in, int t) {
  int b = blockIdx.x;
  int tok = ytr[(b * TT + t) * 3 + 0];
  int par = ytr[(b * TT + t) * 3 + 1];
  for (int k = threadIdx.x; k < 2048; k += 256) {
    float v;
    if (k < 512)       v = emb[(size_t)tok * 512 + k];
    else if (k < 1024) v = states[((size_t)par * BB + b) * DD + (k - 512)];
    else               v = wfeed[b * 1024 + (k - 1024)];
    word_in[b * 2048 + k] = f2bf(v);
    float v2 = (k < 1024) ? v : rfeed[b * 1024 + (k - 1024)];
    rule_in[b * 2560 + k] = f2bf(v2);
  }
}

// ---------------------------------------------------------------------------
// Fused LSTM step: 4 waves per block, wave g computes gate g for 16 columns.
// grid.x = D/16 = 32.  Pointwise gate math + (word) mask-blend fused in-block.
// ---------------------------------------------------------------------------
__global__ void __launch_bounds__(128)
lstm_step(const unsigned short* __restrict__ Xin, int Kin,
          const unsigned short* __restrict__ Hb_old,
          const unsigned short* __restrict__ Wih, const unsigned short* __restrict__ Whh,
          const float* __restrict__ bih, const float* __restrict__ bhh,
          const float* __restrict__ h_old, const float* __restrict__ c_old,
          float* __restrict__ h_new, float* __restrict__ c_new,
          unsigned short* __restrict__ hb_new,
          const int* __restrict__ ytr, int t, int is_word,
          float* __restrict__ states, unsigned short* __restrict__ rule_in,
          unsigned short* __restrict__ inp) {
  __shared__ float gb[4][32][16];
  const int g  = threadIdx.x >> 5;          // gate i/f/g/o
  const int n0 = blockIdx.x * 16;           // column group within D
  v8f acc0 = {}, acc1 = {};
  wmma_tile32(Xin,    Kin, 0, Wih, Kin, g * DD + n0, Kin, acc0, acc1);   // x @ W_ih.T
  wmma_tile32(Hb_old, DD,  0, Whh, DD,  g * DD + n0, DD,  acc0, acc1);   // h @ W_hh.T
  const int lane = threadIdx.x & 31, half = lane >> 4, lrow = lane & 15;
#pragma unroll
  for (int i = 0; i < 8; ++i) {
    gb[g][half * 8 + i][lrow]      = acc0[i];
    gb[g][16 + half * 8 + i][lrow] = acc1[i];
  }
  __syncthreads();
  for (int idx = threadIdx.x; idx < 512; idx += 128) {
    int b = idx >> 4, j = idx & 15, col = n0 + j;
    float iv = gb[0][b][j] + bih[col]            + bhh[col];
    float fv = gb[1][b][j] + bih[DD + col]       + bhh[DD + col];
    float gv = gb[2][b][j] + bih[2 * DD + col]   + bhh[2 * DD + col];
    float ov = gb[3][b][j] + bih[3 * DD + col]   + bhh[3 * DD + col];
    float co = c_old[b * DD + col];
    float c2 = sigm(fv) * co + sigm(iv) * tanhf(gv);
    float h2 = sigm(ov) * tanhf(c2);
    if (is_word) {
      float wm = (float)ytr[(b * TT + t) * 3 + 2];
      h2 = h2 * wm + h_old[b * DD + col] * (1.0f - wm);
      c2 = c2 * wm + co * (1.0f - wm);
    }
    h_new[b * DD + col] = h2;
    c_new[b * DD + col] = c2;
    hb_new[b * DD + col] = f2bf(h2);
    size_t irow = ((size_t)(b * TT + t)) * 3072;
    if (is_word) {
      rule_in[b * 2560 + 2048 + col] = f2bf(h2);   // rule LSTM input tail
      inp[irow + 1536 + col] = f2bf(h2);           // wh_t slot
    } else {
      states[((size_t)(t + 1) * BB + b) * DD + col] = h2;
      inp[irow + col] = f2bf(h2);                  // rh_t slot
    }
  }
}

// ---------------------------------------------------------------------------
// MLP attention (rule: blockIdx.y==0, word: ==1). One block per (b, which).
// ---------------------------------------------------------------------------
__global__ void __launch_bounds__(256)
attn_step(const float* __restrict__ x_enc, const float* __restrict__ x_enc_k,
          const unsigned char* __restrict__ x_mask,
          const float* __restrict__ rh, const float* __restrict__ wh,
          const float* __restrict__ r_aqW, const float* __restrict__ r_aqb,
          const float* __restrict__ r_avW, const float* __restrict__ r_avb,
          const float* __restrict__ w_aqW, const float* __restrict__ w_aqb,
          const float* __restrict__ w_avW, const float* __restrict__ w_avb,
          float* __restrict__ rfeed, float* __restrict__ wfeed,
          unsigned short* __restrict__ inp, int t) {
  __shared__ float qv[512], qw[512], sb[64][4], sv[64];
  __shared__ float mx, sm;
  const int b = blockIdx.x, which = blockIdx.y, tid = threadIdx.x;
  const float* q  = which ? wh    : rh;
  const float* Wq = which ? w_aqW : r_aqW;
  const float* bq = which ? w_aqb : r_aqb;
  const float* wa = which ? w_avW : r_avW;
  const float  ba = which ? w_avb[0] : r_avb[0];
  float* feed = which ? wfeed : rfeed;
  const int base = which ? 2048 : 512;

  for (int k = tid; k < 512; k += 256) qv[k] = q[b * DD + k];
  __syncthreads();
  for (int c = tid; c < 512; c += 256) {                 // qW = q @ Wq.T + bq
    float a = bq[c];
    const float* wr = Wq + (size_t)c * 512;
    for (int k = 0; k < 512; ++k) a += qv[k] * wr[k];
    qw[c] = a;
  }
  __syncthreads();
  {                                                      // scores
    int l = tid >> 2, part = tid & 3;
    const float* kr  = x_enc_k + ((size_t)(b * LXX + l)) * 512 + part * 128;
    const float* qwp = qw + part * 128;
    const float* wap = wa + part * 128;
    float a = 0.0f;
    for (int k = 0; k < 128; ++k) a += tanhf(kr[k] + qwp[k]) * wap[k];
    sb[l][part] = a;
  }
  __syncthreads();
  if (tid < 64) {
    float s = sb[tid][0] + sb[tid][1] + sb[tid][2] + sb[tid][3] + ba;
    if (x_mask[b * LXX + tid]) s = -1.0e9f;
    sv[tid] = s;
  }
  __syncthreads();
  if (tid == 0) { float m = -1.0e30f; for (int i = 0; i < 64; ++i) m = fmaxf(m, sv[i]); mx = m; }
  __syncthreads();
  if (tid < 64) sv[tid] = __expf(sv[tid] - mx);
  __syncthreads();
  if (tid == 0) { float s = 0.0f; for (int i = 0; i < 64; ++i) s += sv[i]; sm = s; }
  __syncthreads();
  for (int c = tid; c < 1024; c += 256) {                // ctx = w @ x_enc
    float a = 0.0f;
    for (int l = 0; l < 64; ++l) a += sv[l] * x_enc[((size_t)(b * LXX + l)) * 1024 + c];
    a /= sm;
    feed[b * 1024 + c] = a;
    inp[((size_t)(b * TT + t)) * 3072 + base + c] = f2bf(a);
  }
}

// ---------------------------------------------------------------------------
// Batched readout, stage 1: hidden = tanh(inp @ (wm ? word_ro : rule_ro).T)
// M=2048 rows, N=512, K=3072. Per-row weight selection by wflag.
// ---------------------------------------------------------------------------
__global__ void __launch_bounds__(128)
readout_hidden(const unsigned short* __restrict__ inp,
               const unsigned short* __restrict__ ruleRo,
               const unsigned short* __restrict__ wordRo,
               const int* __restrict__ ytr, unsigned short* __restrict__ hidden) {
  const int ntile = blockIdx.x * 4 + (threadIdx.x >> 5);
  const int n0 = ntile * 16, m0 = blockIdx.y * 32;
  v8f r0 = {}, r1 = {}, w0 = {}, w1 = {};
  wmma_tile32(inp, 3072, m0, ruleRo, 3072, n0, 3072, r0, r1);
  wmma_tile32(inp, 3072, m0, wordRo, 3072, n0, 3072, w0, w1);
  const int lane = threadIdx.x & 31, half = lane >> 4, lrow = lane & 15;
#pragma unroll
  for (int i = 0; i < 8; ++i) {
    int ra = m0 + half * 8 + i;
    int rb = m0 + 16 + half * 8 + i;
    float wma = (float)ytr[ra * 3 + 2];
    float wmb = (float)ytr[rb * 3 + 2];
    float va = wma != 0.0f ? w0[i] : r0[i];
    float vb = wmb != 0.0f ? w1[i] : r1[i];
    hidden[(size_t)ra * DD + n0 + lrow] = f2bf(tanhf(va));
    hidden[(size_t)rb * DD + n0 + lrow] = f2bf(tanhf(vb));
  }
}

// ---------------------------------------------------------------------------
// Batched readout, stage 2: scores = hidden @ readout_W.T -> d_out (B,T,V)
// M=2048, N=32512 (padded, guarded to 32500), K=512.
// Block = 4 waves -> 64x64 output tile. A panel (64xK) staged in LDS via
// CDNA5 async global->LDS (ASYNCcnt), shared by all 4 waves; each wave runs
// 4 M-subtiles against one B operand register per k-step (4x B amortization).
// ---------------------------------------------------------------------------
__global__ void __launch_bounds__(128)
readout_scores(const unsigned short* __restrict__ hidden,
               const unsigned short* __restrict__ roW, float* __restrict__ out) {
  __shared__ unsigned short sA[64][256];                 // 32 KB A panel chunk
  const int tid  = threadIdx.x;
  const int wave = tid >> 5;
  const int lane = tid & 31, half = lane >> 4, lrow = lane & 15;
  const int ntile = blockIdx.x * 4 + wave;
  const int n0 = ntile * 16;
  const int m0 = blockIdx.y * 64;
  const __bf16* Wb  = reinterpret_cast<const __bf16*>(roW);
  const __bf16* sAb = reinterpret_cast<const __bf16*>(&sA[0][0]);
  v8f acc[4] = {};
  for (int kc = 0; kc < DD; kc += 256) {
    // Stage 64x256 bf16 A chunk: one async b128 per lane per iteration.
    // LDS byte address = low 32 bits of the generic shared-memory address.
#pragma unroll
    for (int j = 0; j < 16; ++j) {
      int q   = j * 128 + tid;            // 2048 b128 transfers per chunk
      int row = q >> 5;
      int col = (q & 31) << 3;            // 8 bf16 per transfer
      unsigned lds_off = (unsigned)(unsigned long long)(const void*)&sA[row][col];
      unsigned long long gaddr =
          (unsigned long long)(const void*)(hidden + (size_t)(m0 + row) * DD + kc + col);
      asm volatile("global_load_async_to_lds_b128 %0, %1, off"
                   :: "v"(lds_off), "v"(gaddr) : "memory");
    }
    asm volatile("s_wait_asynccnt 0x0" ::: "memory");
    __syncthreads();
    for (int k = kc; k < kc + 256; k += 32) {
      v16bf bm;
      const __bf16* wr = Wb + (size_t)(n0 + lrow) * DD + k + half * 8;
#pragma unroll
      for (int j = 0; j < 8; ++j) { bm[j] = wr[j]; bm[8 + j] = wr[16 + j]; }
#pragma unroll
      for (int s = 0; s < 4; ++s) {
        v16bf am;
        const __bf16* ar = sAb + (size_t)(s * 16 + lrow) * 256 + (k - kc) + half * 8;
#pragma unroll
        for (int j = 0; j < 8; ++j) { am[j] = ar[j]; am[8 + j] = ar[16 + j]; }
        acc[s] = __builtin_amdgcn_wmma_f32_16x16x32_bf16(false, am, false, bm,
                                                         (short)0, acc[s], false, false);
      }
    }
    __syncthreads();
  }
  const int col = n0 + lrow;
  if (col < VOC) {
#pragma unroll
    for (int s = 0; s < 4; ++s)
#pragma unroll
      for (int i = 0; i < 8; ++i)
        out[(size_t)(m0 + s * 16 + half * 8 + i) * VOC + col] = acc[s][i];
  }
}

// ---------------------------------------------------------------------------
// Host orchestration
// ---------------------------------------------------------------------------
extern "C" void kernel_launch(void* const* d_in, const int* in_sizes, int n_in,
                              void* d_out, int out_size, void* d_ws, size_t ws_size,
                              hipStream_t stream) {
  const float* x_enc      = (const float*)d_in[0];
  const float* x_enc_k    = (const float*)d_in[1];
  const float* dec_init_h = (const float*)d_in[2];
  const float* dec_init_c = (const float*)d_in[3];
  const unsigned char* x_mask = (const unsigned char*)d_in[4];
  const int*   ytr        = (const int*)d_in[5];
  const float* emb        = (const float*)d_in[6];
  const float* word_W_ih  = (const float*)d_in[7];
  const float* word_W_hh  = (const float*)d_in[8];
  const float* word_b_ih  = (const float*)d_in[9];
  const float* word_b_hh  = (const float*)d_in[10];
  const float* rule_W_ih  = (const float*)d_in[11];
  const float* rule_W_hh  = (const float*)d_in[12];
  const float* rule_b_ih  = (const float*)d_in[13];
  const float* rule_b_hh  = (const float*)d_in[14];
  const float* rule_aq_W  = (const float*)d_in[15];
  const float* rule_aq_b  = (const float*)d_in[16];
  const float* rule_av_W  = (const float*)d_in[17];
  const float* rule_av_b  = (const float*)d_in[18];
  const float* word_aq_W  = (const float*)d_in[19];
  const float* word_aq_b  = (const float*)d_in[20];
  const float* word_av_W  = (const float*)d_in[21];
  const float* word_av_b  = (const float*)d_in[22];
  const float* rule_ro_W  = (const float*)d_in[23];
  const float* word_ro_W  = (const float*)d_in[24];
  const float* readout_W  = (const float*)d_in[25];
  float* out = (float*)d_out;

  char* ws = (char*)d_ws;
  size_t off = 0;
  auto take = [&](size_t bytes) -> void* {
    void* p = ws + off;
    off += (bytes + 255) & ~(size_t)255;
    return p;
  };
  unsigned short* wWih   = (unsigned short*)take((size_t)2048 * 2048 * 2);
  unsigned short* wWhh   = (unsigned short*)take((size_t)2048 * 512 * 2);
  unsigned short* rWih   = (unsigned short*)take((size_t)2048 * 2560 * 2);
  unsigned short* rWhh   = (unsigned short*)take((size_t)2048 * 512 * 2);
  unsigned short* ruleRo = (unsigned short*)take((size_t)512 * 3072 * 2);
  unsigned short* wordRo = (unsigned short*)take((size_t)512 * 3072 * 2);
  unsigned short* roW    = (unsigned short*)take((size_t)VPAD * 512 * 2);
  float* states = (float*)take((size_t)(TT + 1) * BB * DD * 4);
  float* whb2[2]; float* wcb2[2]; float* rhb2[2]; float* rcb2[2];
  unsigned short* whbf[2]; unsigned short* rhbf[2];
  for (int p = 0; p < 2; ++p) {
    whb2[p] = (float*)take((size_t)BB * DD * 4);
    wcb2[p] = (float*)take((size_t)BB * DD * 4);
    rhb2[p] = (float*)take((size_t)BB * DD * 4);
    rcb2[p] = (float*)take((size_t)BB * DD * 4);
    whbf[p] = (unsigned short*)take((size_t)BB * DD * 2);
    rhbf[p] = (unsigned short*)take((size_t)BB * DD * 2);
  }
  float* rfeed = (float*)take((size_t)BB * 1024 * 4);
  float* wfeed = (float*)take((size_t)BB * 1024 * 4);
  unsigned short* word_in = (unsigned short*)take((size_t)BB * 2048 * 2);
  unsigned short* rule_in = (unsigned short*)take((size_t)BB * 2560 * 2);
  unsigned short* inp     = (unsigned short*)take((size_t)BB * TT * 3072 * 2);
  unsigned short* hidden  = (unsigned short*)take((size_t)BB * TT * DD * 2);

  // One-shot weight conversion fp32 -> bf16 (deterministic, redone per call)
  cvt_bf16_pad<<<2048, 256, 0, stream>>>(word_W_ih, wWih,   (long)2048 * 2048, (long)2048 * 2048);
  cvt_bf16_pad<<<1024, 256, 0, stream>>>(word_W_hh, wWhh,   (long)2048 * 512,  (long)2048 * 512);
  cvt_bf16_pad<<<2048, 256, 0, stream>>>(rule_W_ih, rWih,   (long)2048 * 2560, (long)2048 * 2560);
  cvt_bf16_pad<<<1024, 256, 0, stream>>>(rule_W_hh, rWhh,   (long)2048 * 512,  (long)2048 * 512);
  cvt_bf16_pad<<<1024, 256, 0, stream>>>(rule_ro_W, ruleRo, (long)512 * 3072,  (long)512 * 3072);
  cvt_bf16_pad<<<1024, 256, 0, stream>>>(word_ro_W, wordRo, (long)512 * 3072,  (long)512 * 3072);
  cvt_bf16_pad<<<4096, 256, 0, stream>>>(readout_W, roW,    (long)VOC * 512,   (long)VPAD * 512);

  // State init: states0 = 0, feeds = 0, h/c <- dec_init
  zero_f32<<<1024, 256, 0, stream>>>(states, (long)(TT + 1) * BB * DD);
  zero_f32<<<64,   256, 0, stream>>>(rfeed,  (long)BB * 1024);
  zero_f32<<<64,   256, 0, stream>>>(wfeed,  (long)BB * 1024);
  copy_hc<<<64, 256, 0, stream>>>(dec_init_h, whb2[0], whbf[0], BB * DD);
  copy_hc<<<64, 256, 0, stream>>>(dec_init_h, rhb2[0], rhbf[0], BB * DD);
  copy_hc<<<64, 256, 0, stream>>>(dec_init_c, wcb2[0], (unsigned short*)nullptr, BB * DD);
  copy_hc<<<64, 256, 0, stream>>>(dec_init_c, rcb2[0], (unsigned short*)nullptr, BB * DD);

  // Sequential scan (latency-bound; weights resident in 192MB L2)
  for (int t = 0; t < TT; ++t) {
    const int p = t & 1, q = 1 - p;
    prep_step<<<BB, 256, 0, stream>>>(ytr, emb, states, rfeed, wfeed, word_in, rule_in, t);
    lstm_step<<<DD / 16, 128, 0, stream>>>(word_in, 2048, whbf[p], wWih, wWhh,
                                           word_b_ih, word_b_hh,
                                           whb2[p], wcb2[p], whb2[q], wcb2[q], whbf[q],
                                           ytr, t, 1, states, rule_in, inp);
    lstm_step<<<DD / 16, 128, 0, stream>>>(rule_in, 2560, rhbf[p], rWih, rWhh,
                                           rule_b_ih, rule_b_hh,
                                           rhb2[p], rcb2[p], rhb2[q], rcb2[q], rhbf[q],
                                           ytr, t, 0, states, rule_in, inp);
    attn_step<<<dim3(BB, 2), 256, 0, stream>>>(x_enc, x_enc_k, x_mask,
                                               rhb2[q], whb2[q],
                                               rule_aq_W, rule_aq_b, rule_av_W, rule_av_b,
                                               word_aq_W, word_aq_b, word_av_W, word_av_b,
                                               rfeed, wfeed, inp, t);
  }

  // Batched readout: stage1 then L2-reuse-blocked score GEMM into d_out
  readout_hidden<<<dim3(DD / 16 / 4, BB * TT / 32), 128, 0, stream>>>(inp, ruleRo, wordRo, ytr, hidden);
  readout_scores<<<dim3(VPAD / 64, BB * TT / 64), 128, 0, stream>>>(hidden, roW, out);
}
